// Net_747324309710
// MI455X (gfx1250) — compile-verified, matrix operand order checked by reading, not run on hardware
//
#include <hip/hip_runtime.h>

typedef _Float16 f16;
typedef __attribute__((ext_vector_type(16))) _Float16 v16h;
typedef __attribute__((ext_vector_type(8)))  _Float16 v8h;
typedef __attribute__((ext_vector_type(4)))  _Float16 v4h;
typedef __attribute__((ext_vector_type(8)))  float    v8f;

#define N_NODES 50000
#define IN_DIM  7688
#define HID     200
#define OUT_DIM 8
#define N_EDGES 1600000

#define KPAD1 7712   // 241 * 32
#define NPAD1 208    // 13 * 16
#define KTILES1 (KPAD1 / 32)
#define KPAD2 224    // 7 * 32
#define NPAD2 16

#define G1_ROWS  64
#define G1_WAVES 13
#define A_STRIDE 40  // halfs; 80B rows keep 16B fragment alignment, spread banks

// ---------------- prep kernels ----------------

__global__ void k_fill1(float* p, int n) {
    int i = blockIdx.x * 256 + threadIdx.x;
    if (i < n) p[i] = 1.0f;   // self-loop weight baked into degree
}

__global__ void k_deg_scatter(const int* __restrict__ dst,
                              const float* __restrict__ ew,
                              float* __restrict__ deg) {
    int e = blockIdx.x * 256 + threadIdx.x;
    if (e < N_EDGES) atomicAdd(&deg[dst[e]], ew[e]);
}

__global__ void k_dis(float* deg) {
    int i = blockIdx.x * 256 + threadIdx.x;
    if (i < N_NODES) {
        float d = deg[i];
        deg[i] = (d > 0.0f) ? rsqrtf(d) : 0.0f;   // deg -> D^-1/2 in place
    }
}

__global__ void k_norm(const int* __restrict__ src, const int* __restrict__ dst,
                       const float* __restrict__ ew, const float* __restrict__ dis,
                       float* __restrict__ norm) {
    int e = blockIdx.x * 256 + threadIdx.x;
    if (e < N_EDGES) norm[e] = dis[src[e]] * ew[e] * dis[dst[e]];
}

// W1 [IN_DIM][HID] f32 -> Wt1 [NPAD1][KPAD1] f16 (transposed, zero padded); grid.y = n
__global__ void k_prep_w1(const float* __restrict__ W1, f16* __restrict__ Wt1) {
    int k = blockIdx.x * 256 + threadIdx.x;
    int n = blockIdx.y;
    if (k < KPAD1) {
        float v = (n < HID && k < IN_DIM) ? W1[(long)k * HID + n] : 0.0f;
        Wt1[(long)n * KPAD1 + k] = (f16)v;
    }
}

// W2 [HID][OUT] -> Wt2 [NPAD2][KPAD2] f16 (transposed, zero padded)
__global__ void k_prep_w2(const float* __restrict__ W2, f16* __restrict__ Wt2) {
    int i = blockIdx.x * 256 + threadIdx.x;
    if (i < NPAD2 * KPAD2) {
        int n = i / KPAD2;
        int k = i % KPAD2;
        float v = (n < OUT_DIM && k < HID) ? W2[k * OUT_DIM + n] : 0.0f;
        Wt2[i] = (f16)v;
    }
}

// ---------------- GEMM 1: h1 = X @ W1 (f16 WMMA, f32 accumulate) ----------------

// Stage one [64 x 32] f32 tile of X into f16 LDS via b128 loads + b64 LDS stores.
template<bool GUARD>
__device__ __forceinline__ void stage_tile(f16* __restrict__ buf,
                                           const float* __restrict__ X,
                                           int m0, int k0, int tid) {
    #pragma unroll
    for (int idx = tid; idx < G1_ROWS * 8; idx += G1_WAVES * 32) {
        const int r  = idx >> 3;
        const int kc = (idx & 7) << 2;          // 4-float chunk, 16B aligned
        const float* gp = X + (long)(m0 + r) * IN_DIM + (k0 + kc);
        float4 v = make_float4(0.f, 0.f, 0.f, 0.f);
        if (!GUARD) {
            v = *(const float4*)gp;
            __builtin_prefetch(gp + 64, 0, 1);  // pull K-tile t+2 toward the WGP
        } else if ((m0 + r) < N_NODES && (k0 + kc) < IN_DIM) {
            v = *(const float4*)gp;             // IN_DIM % 4 == 0 -> chunk all-or-nothing
        }
        v4h h = { (f16)v.x, (f16)v.y, (f16)v.z, (f16)v.w };
        *(v4h*)&buf[r * A_STRIDE + kc] = h;     // 8B aligned ds_store_b64
    }
}

__launch_bounds__(G1_WAVES * 32)
__global__ void k_gemm1(const float* __restrict__ X,
                        const f16* __restrict__ Wt1,
                        float* __restrict__ h1) {
    __shared__ f16 At[2][G1_ROWS * A_STRIDE];   // double buffered, 2 x 5KB

    const int tid   = threadIdx.x;
    const int wave  = tid >> 5;        // col tile 0..12
    const int lane  = tid & 31;
    const int lrow  = lane & 15;
    const int halfk = (lane >> 4) * 8; // 0 or 8 (16-bit A/B VGPR layout)
    const int m0    = blockIdx.x * G1_ROWS;
    const bool interior = (m0 + G1_ROWS <= N_NODES);

    v8f acc0 = {}, acc1 = {}, acc2 = {}, acc3 = {};
    const f16* bcol = Wt1 + (long)(wave * 16 + lrow) * KPAD1 + halfk;

    if (interior) stage_tile<false>(At[0], X, m0, 0, tid);
    else          stage_tile<true >(At[0], X, m0, 0, tid);
    __syncthreads();

    int p = 0;
    for (int t = 0; t < KTILES1; ++t) {
        const int k0 = t * 32;
        // stage next tile into the other buffer while we compute this one
        if (t + 1 < KTILES1) {
            const int kn = k0 + 32;
            const bool g = (!interior) || (kn + 32 > IN_DIM);   // only last K-tile
            if (g) stage_tile<true >(At[p ^ 1], X, m0, kn, tid);
            else   stage_tile<false>(At[p ^ 1], X, m0, kn, tid);
        }

        // B fragment: two contiguous 16B loads from L2-resident transposed W1
        union { v16h v; v8h h[2]; } b;
        const f16* bp = bcol + k0;
        b.h[0] = *(const v8h*)bp;
        b.h[1] = *(const v8h*)(bp + 16);

        const f16* abase = At[p];
#define DO_RT(RT, ACC) {                                                        \
        union { v16h v; v8h h[2]; } a;                                          \
        const f16* ap = abase + ((RT) * 16 + lrow) * A_STRIDE + halfk;          \
        a.h[0] = *(const v8h*)ap;                                               \
        a.h[1] = *(const v8h*)(ap + 16);                                        \
        ACC = __builtin_amdgcn_wmma_f32_16x16x32_f16(false, a.v, false, b.v,    \
                                                     (short)0, ACC, false, false); }
        DO_RT(0, acc0)
        DO_RT(1, acc1)
        DO_RT(2, acc2)
        DO_RT(3, acc3)
#undef DO_RT
        __syncthreads();
        p ^= 1;
    }

    // C/D layout: element i, lanes 0-15 -> row i, lanes 16-31 -> row i+8; col = lane%16
    const int col = wave * 16 + lrow;
    if (col < HID) {
        if (interior) {
            #pragma unroll
            for (int i = 0; i < 8; ++i) {
                int rr = (lane < 16) ? i : (i + 8);
                h1[(long)(m0 + rr)      * HID + col] = acc0[i];
                h1[(long)(m0 + 16 + rr) * HID + col] = acc1[i];
                h1[(long)(m0 + 32 + rr) * HID + col] = acc2[i];
                h1[(long)(m0 + 48 + rr) * HID + col] = acc3[i];
            }
        } else {
            #pragma unroll
            for (int i = 0; i < 8; ++i) {
                int rr = (lane < 16) ? i : (i + 8);
                int r0 = m0 + rr, r1 = m0 + 16 + rr, r2 = m0 + 32 + rr, r3 = m0 + 48 + rr;
                if (r0 < N_NODES) h1[(long)r0 * HID + col] = acc0[i];
                if (r1 < N_NODES) h1[(long)r1 * HID + col] = acc1[i];
                if (r2 < N_NODES) h1[(long)r2 * HID + col] = acc2[i];
                if (r3 < N_NODES) h1[(long)r3 * HID + col] = acc3[i];
            }
        }
    }
}

// ---------------- message passing, layer 1 ----------------

// a1 = dis[i]^2 * h1[i]  (self-loop term of segment_sum)
__global__ void k_self1(const float* __restrict__ h1, const float* __restrict__ dis,
                        float* __restrict__ a1) {
    long i = (long)blockIdx.x * 256 + threadIdx.x;
    if (i < (long)N_NODES * HID) {
        float d = dis[i / HID];
        a1[i] = h1[i] * d * d;
    }
}

// one edge per block, channel per thread
__launch_bounds__(224)
__global__ void k_scat1(const int* __restrict__ src, const int* __restrict__ dst,
                        const float* __restrict__ norm, const float* __restrict__ h1,
                        float* __restrict__ a1) {
    int e = blockIdx.x;
    int c = threadIdx.x;
    if (c < HID) {
        int s = src[e], d = dst[e];
        float w = norm[e];
        atomicAdd(&a1[(long)d * HID + c], h1[(long)s * HID + c] * w);
    }
}

__global__ void k_bias_relu(float* __restrict__ a1, const float* __restrict__ b1) {
    long i = (long)blockIdx.x * 256 + threadIdx.x;
    if (i < (long)N_NODES * HID) {
        float v = a1[i] + b1[i % HID];
        a1[i] = v > 0.0f ? v : 0.0f;
    }
}

// ---------------- GEMM 2: h2 = a1 @ W2 (one N tile per wave) ----------------

__launch_bounds__(256)
__global__ void k_gemm2(const float* __restrict__ a1,
                        const f16* __restrict__ Wt2,
                        float* __restrict__ h2) {
    const int wave = threadIdx.x >> 5;
    const int lane = threadIdx.x & 31;
    const int mt   = blockIdx.x * 8 + wave;      // 16-row tile index (3125 exact tiles)
    if (mt >= N_NODES / 16) return;              // wave-uniform exit, EXEC stays full

    const int lrow  = lane & 15;
    const int halfk = (lane >> 4) * 8;
    const int row   = mt * 16 + lrow;

    v8f acc = {};
    const f16* bbase = Wt2 + lrow * KPAD2 + halfk;
    const float* arow = a1 + (long)row * HID;

    for (int k0 = 0; k0 < KPAD2; k0 += 32) {
        union { v16h v; v8h h[2]; } a, b;
        b.h[0] = *(const v8h*)(bbase + k0);
        b.h[1] = *(const v8h*)(bbase + k0 + 16);

        const int ka = k0 + halfk;               // multiple of 8; HID % 8 == 0
        const int kb = ka + 16;
        float4 x0 = make_float4(0,0,0,0), x1 = x0, x2 = x0, x3 = x0;
        if (ka < HID) { x0 = *(const float4*)(arow + ka); x1 = *(const float4*)(arow + ka + 4); }
        if (kb < HID) { x2 = *(const float4*)(arow + kb); x3 = *(const float4*)(arow + kb + 4); }
        a.h[0] = v8h{ (f16)x0.x, (f16)x0.y, (f16)x0.z, (f16)x0.w,
                      (f16)x1.x, (f16)x1.y, (f16)x1.z, (f16)x1.w };
        a.h[1] = v8h{ (f16)x2.x, (f16)x2.y, (f16)x2.z, (f16)x2.w,
                      (f16)x3.x, (f16)x3.y, (f16)x3.z, (f16)x3.w };

        acc = __builtin_amdgcn_wmma_f32_16x16x32_f16(false, a.v, false, b.v,
                                                     (short)0, acc, false, false);
    }

    if (lrow < OUT_DIM) {
        #pragma unroll
        for (int i = 0; i < 8; ++i) {
            int r = mt * 16 + ((lane < 16) ? i : (i + 8));
            h2[(long)r * OUT_DIM + lrow] = acc[i];
        }
    }
}

// ---------------- message passing, layer 2 (writes d_out) ----------------

// out = dis^2 * h2 + b2   (self-loop + bias; fully initializes d_out)
__global__ void k_self2(const float* __restrict__ h2, const float* __restrict__ dis,
                        const float* __restrict__ b2, float* __restrict__ out) {
    long i = (long)blockIdx.x * 256 + threadIdx.x;
    if (i < (long)N_NODES * OUT_DIM) {
        float d = dis[i >> 3];
        out[i] = h2[i] * d * d + b2[i & 7];
    }
}

__global__ void k_scat2(const int* __restrict__ src, const int* __restrict__ dst,
                        const float* __restrict__ norm, const float* __restrict__ h2,
                        float* __restrict__ out) {
    long i = (long)blockIdx.x * 256 + threadIdx.x;
    if (i < (long)N_EDGES * OUT_DIM) {
        long e = i >> 3;
        int  c = (int)(i & 7);
        atomicAdd(&out[(long)dst[e] * OUT_DIM + c],
                  h2[(long)src[e] * OUT_DIM + c] * norm[e]);
    }
}

// ---------------- host side ----------------

extern "C" void kernel_launch(void* const* d_in, const int* in_sizes, int n_in,
                              void* d_out, int out_size, void* d_ws, size_t ws_size,
                              hipStream_t stream) {
    (void)in_sizes; (void)n_in; (void)out_size; (void)ws_size;

    const float* X  = (const float*)d_in[0];
    const float* ew = (const float*)d_in[1];
    const float* W1 = (const float*)d_in[2];
    const float* b1 = (const float*)d_in[3];
    const float* W2 = (const float*)d_in[4];
    const float* b2 = (const float*)d_in[5];
    const int*   ei = (const int*)d_in[6];
    const int* src = ei;
    const int* dst = ei + N_EDGES;
    float* out = (float*)d_out;

    char* ws = (char*)d_ws;
    size_t off = 0;
    auto carve = [&](size_t bytes) -> char* {
        char* p = ws + off;
        off = (off + bytes + 255) & ~(size_t)255;
        return p;
    };
    float* deg  = (float*)carve(sizeof(float) * N_NODES);           // becomes dis
    float* norm = (float*)carve(sizeof(float) * N_EDGES);
    float* h1   = (float*)carve(sizeof(float) * (size_t)N_NODES * HID);
    float* a1   = (float*)carve(sizeof(float) * (size_t)N_NODES * HID);
    float* h2   = (float*)carve(sizeof(float) * (size_t)N_NODES * OUT_DIM);
    f16*   Wt1  = (f16*)carve(sizeof(f16) * (size_t)NPAD1 * KPAD1);
    f16*   Wt2  = (f16*)carve(sizeof(f16) * (size_t)NPAD2 * KPAD2);

    // normalization
    k_fill1<<<(N_NODES + 255) / 256, 256, 0, stream>>>(deg, N_NODES);
    k_deg_scatter<<<(N_EDGES + 255) / 256, 256, 0, stream>>>(dst, ew, deg);
    k_dis<<<(N_NODES + 255) / 256, 256, 0, stream>>>(deg);
    k_norm<<<(N_EDGES + 255) / 256, 256, 0, stream>>>(src, dst, ew, deg, norm);

    // weight transpose/pad to f16
    {
        dim3 g1((KPAD1 + 255) / 256, NPAD1);
        k_prep_w1<<<g1, 256, 0, stream>>>(W1, Wt1);
        k_prep_w2<<<(NPAD2 * KPAD2 + 255) / 256, 256, 0, stream>>>(W2, Wt2);
    }

    // layer 1
    k_gemm1<<<(N_NODES + G1_ROWS - 1) / G1_ROWS, G1_WAVES * 32, 0, stream>>>(X, Wt1, h1);
    {
        long n = (long)N_NODES * HID;
        k_self1<<<(unsigned)((n + 255) / 256), 256, 0, stream>>>(h1, deg, a1);
    }
    k_scat1<<<N_EDGES, 224, 0, stream>>>(src, dst, norm, h1, a1);
    {
        long n = (long)N_NODES * HID;
        k_bias_relu<<<(unsigned)((n + 255) / 256), 256, 0, stream>>>(a1, b1);
    }

    // layer 2
    k_gemm2<<<(N_NODES / 16 + 7) / 8, 256, 0, stream>>>(a1, Wt2, h2);
    {
        long n = (long)N_NODES * OUT_DIM;
        k_self2<<<(unsigned)((n + 255) / 256), 256, 0, stream>>>(h2, deg, b2, out);
        long m = (long)N_EDGES * OUT_DIM;
        k_scat2<<<(unsigned)((m + 255) / 256), 256, 0, stream>>>(src, dst, norm, h2, out);
    }
}
